// CTEBlock_35407710388553
// MI455X (gfx1250) — matrix operand order repaired and checked by reading, not compiled, not run on hardware
//
#include <hip/hip_runtime.h>
#include <math.h>

// ---------------------------------------------------------------------------
// CDNA5 (gfx1250) transformer block with candidate elimination.
// All GEMMs via v_wmma_f32_16x16x32_bf16, wave32, 128x128x32 block tiles,
// double-buffered LDS staging (async global->LDS when available).
// ---------------------------------------------------------------------------

typedef __bf16 bf16_t;
typedef bf16_t bfx16 __attribute__((ext_vector_type(16)));
typedef bf16_t bfx8  __attribute__((ext_vector_type(8)));
typedef float  fx8   __attribute__((ext_vector_type(8)));
typedef int    v4i   __attribute__((ext_vector_type(4)));

#define BM 128
#define BN 128
#define BK 32

enum { EPI_PLAIN = 0, EPI_GELU = 1, EPI_RES = 2 };

#if defined(__gfx1250__) && __has_builtin(__builtin_amdgcn_global_load_async_to_lds_b128)
#define ASYNC_LDS 1
#else
#define ASYNC_LDS 0
#endif

#if ASYNC_LDS
typedef __attribute__((address_space(1))) v4i* gv4i_p;  // global v4i*
typedef __attribute__((address_space(3))) v4i* lv4i_p;  // LDS v4i*

__device__ __forceinline__ void async_cp16(const bf16_t* g, bf16_t* l) {
  // Generic->AS1 is value-preserving; generic->AS3 is addr[31:0] per ISA.
  __builtin_amdgcn_global_load_async_to_lds_b128(
      (gv4i_p)(unsigned long long)g, (lv4i_p)(unsigned long long)l, 0, 0);
}
__device__ __forceinline__ void wait_async0() {
#if __has_builtin(__builtin_amdgcn_s_wait_asynccnt)
  __builtin_amdgcn_s_wait_asynccnt(0);
#else
  asm volatile("s_wait_asynccnt 0x0" ::: "memory");
#endif
}
#endif

__device__ __forceinline__ bfx16 join8(bfx8 lo, bfx8 hi) {
  bfx16 r;
#pragma unroll
  for (int i = 0; i < 8; ++i) { r[i] = lo[i]; r[i + 8] = hi[i]; }
  return r;
}

__device__ __forceinline__ float gelu_exact(float x) {
  return 0.5f * x * (1.0f + erff(x * 0.70710678118654752440f));
}

// Generic bf16 WMMA GEMM: C = epi(alpha * A@B + bias [+ res])
// A: MxK row-major (lda). B: KxN row-major (ldb) if !BT, else B src is NxK
// row-major and we consume its transpose. Batch offset: z1=z/bdiv, z2=z%bdiv.
// Out-of-range A rows / B cols may hold stale LDS data: C[m,n] depends only on
// A row m and B col n, and OOB (m,n) are masked at the store, so it is safe.
template <int EPI, bool BT, typename OutT>
__global__ __launch_bounds__(256) void gemm_wmma_bf16(
    const bf16_t* __restrict__ A, const bf16_t* __restrict__ Bm,
    const float* __restrict__ bias, const float* __restrict__ res,
    OutT* __restrict__ C, int M, int Nc, int K, int lda, int ldb, int ldc,
    int bdiv, long aS1, long aS2, long bS1, long bS2, long cS1, long cS2,
    float alpha) {
  __shared__ bf16_t sA[2][BM * BK];
  __shared__ bf16_t sB[2][BK * BN];

  const int z = blockIdx.z;
  const int z1 = z / bdiv, z2 = z % bdiv;
  A += z1 * aS1 + z2 * aS2;
  Bm += z1 * bS1 + z2 * bS2;
  const long coff = z1 * cS1 + z2 * cS2;
  C += coff;
  if (EPI == EPI_RES) res += coff;

  const int t = threadIdx.x;
  const int lane = t & 31;
  const int wv = t >> 5;                // 8 waves
  const int wm0 = (wv >> 1) * 32;       // 0,32,64,96
  const int wn0 = (wv & 1) * 64;        // 0,64
  const int half = lane >> 4;
  const int lrow = lane & 15;

  const int m0 = blockIdx.y * BM;
  const int n0 = blockIdx.x * BN;

  fx8 acc[2][4];
#pragma unroll
  for (int i = 0; i < 2; ++i)
#pragma unroll
    for (int j = 0; j < 4; ++j) acc[i][j] = (fx8)0.0f;

  // Staging index precompute (256 threads, 16 bf16 each per tile).
  const int ar = t >> 1;             // A: row within tile
  const int ac = (t & 1) * 16;       // A: k chunk
  const int bk = t >> 3;             // B: k row
  const int bc = (t & 7) * 16;       // B: n chunk
  const int tn = t >> 1;             // BT: source row (N index)
  const int tc = (t & 1) * 16;       // BT: k chunk

  // Register-path staging state.
  uint4 ra0, ra1, rb0, rb1;
  union { uint4 u[2]; bf16_t h[16]; } rbt;

  auto loadT = [&](int kk) {
    ra0 = make_uint4(0, 0, 0, 0); ra1 = ra0;
    const int gm = m0 + ar;
    if (gm < M) {
      const bf16_t* p = A + (long)gm * lda + kk + ac;
      ra0 = *(const uint4*)p;
      ra1 = *(const uint4*)(p + 8);
      __builtin_prefetch(p + 2 * BK, 0, 3);
    }
    if (!BT) {
      rb0 = make_uint4(0, 0, 0, 0); rb1 = rb0;
      const int gn = n0 + bc;
      if (gn < Nc) {
        const bf16_t* p = Bm + (long)(kk + bk) * ldb + gn;
        rb0 = *(const uint4*)p;
        rb1 = *(const uint4*)(p + 8);
      }
    } else {
      rbt.u[0] = make_uint4(0, 0, 0, 0); rbt.u[1] = rbt.u[0];
      const int gn = n0 + tn;
      if (gn < Nc) {
        const bf16_t* p = Bm + (long)gn * ldb + kk + tc;
        rbt.u[0] = *(const uint4*)p;
        rbt.u[1] = *(const uint4*)(p + 8);
      }
    }
  };
  auto storeT = [&](int buf) {
    *(uint4*)&sA[buf][ar * BK + ac] = ra0;
    *(uint4*)&sA[buf][ar * BK + ac + 8] = ra1;
    if (!BT) {
      *(uint4*)&sB[buf][bk * BN + bc] = rb0;
      *(uint4*)&sB[buf][bk * BN + bc + 8] = rb1;
    } else {
#pragma unroll
      for (int i = 0; i < 16; ++i) sB[buf][(tc + i) * BN + tn] = rbt.h[i];
    }
  };

#if ASYNC_LDS
  auto stageAsync = [&](int kk, int buf) {
    const int gm = m0 + ar;
    if (gm < M) {
      const bf16_t* p = A + (long)gm * lda + kk + ac;
      bf16_t* l = &sA[buf][ar * BK + ac];
      async_cp16(p, l);
      async_cp16(p + 8, l + 8);
    }
    if (!BT) {
      const int gn = n0 + bc;
      if (gn < Nc) {
        const bf16_t* p = Bm + (long)(kk + bk) * ldb + gn;
        bf16_t* l = &sB[buf][bk * BN + bc];
        async_cp16(p, l);
        async_cp16(p + 8, l + 8);
      }
    } else {  // transpose scatter stays on the register path
      rbt.u[0] = make_uint4(0, 0, 0, 0); rbt.u[1] = rbt.u[0];
      const int gn = n0 + tn;
      if (gn < Nc) {
        const bf16_t* p = Bm + (long)gn * ldb + kk + tc;
        rbt.u[0] = *(const uint4*)p;
        rbt.u[1] = *(const uint4*)(p + 8);
      }
#pragma unroll
      for (int i = 0; i < 16; ++i) sB[buf][(tc + i) * BN + tn] = rbt.h[i];
    }
  };
#endif

  const int nt = K / BK;  // K is always a multiple of 32 in this workload.

  // Prologue: tile 0 -> buffer 0.
#if ASYNC_LDS
  stageAsync(0, 0);
  wait_async0();
#else
  loadT(0);
  storeT(0);
#endif
  __syncthreads();

  for (int it = 0; it < nt; ++it) {
    const int cur = it & 1;
    const bool more = (it + 1) < nt;

#if ASYNC_LDS
    if (more) stageAsync((it + 1) * BK, cur ^ 1);
#else
    if (more) loadT((it + 1) * BK);
#endif

    // A fragment: lane (half,lrow): elems 0-7 -> k=half*8.., elems 8-15 -> k=16+half*8..
    bfx16 afrag[2];
#pragma unroll
    for (int i = 0; i < 2; ++i) {
      const bf16_t* p = &sA[cur][(wm0 + i * 16 + lrow) * BK + half * 8];
      afrag[i] = join8(*(const bfx8*)p, *(const bfx8*)(p + 16));
    }
    // B fragment: lane holds n-col lrow, elems 0-7 -> k=half*8.., elems 8-15 -> k=16+half*8..
    bfx16 bfrag[4];
#pragma unroll
    for (int j = 0; j < 4; ++j) {
      const bf16_t* p = &sB[cur][wn0 + j * 16 + lrow];
#pragma unroll
      for (int s = 0; s < 8; ++s) {
        bfrag[j][s] = p[(half * 8 + s) * BN];
        bfrag[j][s + 8] = p[(16 + half * 8 + s) * BN];
      }
    }
#pragma unroll
    for (int i = 0; i < 2; ++i)
#pragma unroll
      for (int j = 0; j < 4; ++j)
        acc[i][j] = __builtin_amdgcn_wmma_f32_16x16x32_bf16(
            false, afrag[i], false, bfrag[j], (short)0, acc[i][j], false, false);

#if ASYNC_LDS
    if (more) wait_async0();
#else
    if (more) storeT(cur ^ 1);
#endif
    __syncthreads();
  }

  // Epilogue. C/D layout: elem r -> M = r + 8*half, N = lrow.
#pragma unroll
  for (int i = 0; i < 2; ++i) {
#pragma unroll
    for (int j = 0; j < 4; ++j) {
#pragma unroll
      for (int r = 0; r < 8; ++r) {
        const int gm = m0 + wm0 + i * 16 + half * 8 + r;
        const int gn = n0 + wn0 + j * 16 + lrow;
        if (gm < M && gn < Nc) {
          float v = acc[i][j][r] * alpha;
          if (bias) v += bias[gn];
          if (EPI == EPI_GELU) v = gelu_exact(v);
          if (EPI == EPI_RES) v += res[(long)gm * ldc + gn];
          C[(long)gm * ldc + gn] = (OutT)v;
        }
      }
    }
  }
}

// ---------------------------------------------------------------------------

__global__ void cvt_f32_bf16(const float* __restrict__ s, bf16_t* __restrict__ d,
                             long n) {
  long i = (long)blockIdx.x * blockDim.x + threadIdx.x;
  if (i < n) d[i] = (bf16_t)s[i];
}

__global__ __launch_bounds__(256) void layernorm_bf16(
    const float* __restrict__ x, const float* __restrict__ g,
    const float* __restrict__ b, bf16_t* __restrict__ out) {
  __shared__ float red[256];
  const long row = blockIdx.x;
  const float* p = x + row * 768;
  const int t = threadIdx.x;
  const float v0 = p[t], v1 = p[t + 256], v2 = p[t + 512];
  red[t] = v0 + v1 + v2;
  __syncthreads();
  for (int s = 128; s > 0; s >>= 1) {
    if (t < s) red[t] += red[t + s];
    __syncthreads();
  }
  const float mean = red[0] * (1.0f / 768.0f);
  __syncthreads();
  const float d0 = v0 - mean, d1 = v1 - mean, d2 = v2 - mean;
  red[t] = d0 * d0 + d1 * d1 + d2 * d2;
  __syncthreads();
  for (int s = 128; s > 0; s >>= 1) {
    if (t < s) red[t] += red[t + s];
    __syncthreads();
  }
  const float rstd = rsqrtf(red[0] * (1.0f / 768.0f) + 1e-5f);
  bf16_t* o = out + row * 768;
  o[t] = (bf16_t)(d0 * rstd * g[t] + b[t]);
  o[t + 256] = (bf16_t)(d1 * rstd * g[t + 256] + b[t + 256]);
  o[t + 512] = (bf16_t)(d2 * rstd * g[t + 512] + b[t + 512]);
}

__global__ __launch_bounds__(128) void softmax_row(float* __restrict__ corr,
                                                   bf16_t* __restrict__ abf) {
  __shared__ float red[128];
  const long row = blockIdx.x;
  float* p = corr + row * 320;
  const int t = threadIdx.x;
  const float v0 = p[t], v1 = p[t + 128];
  const float v2 = (t < 64) ? p[t + 256] : -3.0e38f;
  red[t] = fmaxf(v0, fmaxf(v1, v2));
  __syncthreads();
  for (int s = 64; s > 0; s >>= 1) {
    if (t < s) red[t] = fmaxf(red[t], red[t + s]);
    __syncthreads();
  }
  const float mx = red[0];
  __syncthreads();
  const float e0 = __expf(v0 - mx), e1 = __expf(v1 - mx);
  const float e2 = (t < 64) ? __expf(v2 - mx) : 0.0f;
  red[t] = e0 + e1 + e2;
  __syncthreads();
  for (int s = 64; s > 0; s >>= 1) {
    if (t < s) red[t] += red[t + s];
    __syncthreads();
  }
  const float inv = 1.0f / red[0];
  bf16_t* o = abf + row * 320;
  p[t] = e0 * inv;
  o[t] = (bf16_t)(e0 * inv);
  p[t + 128] = e1 * inv;
  o[t + 128] = (bf16_t)(e1 * inv);
  if (t < 64) {
    p[t + 256] = e2 * inv;
    o[t + 256] = (bf16_t)(e2 * inv);
  }
}

__global__ __launch_bounds__(256) void ce_score(const float* __restrict__ cr,
                                                const float* __restrict__ ct,
                                                float* __restrict__ score) {
  const int b = blockIdx.x, s = threadIdx.x;
  float acc = 0.0f;
  const long base0 = (long)b * 12 * 320 * 320 + 64 + s;
  for (int h = 0; h < 12; ++h) {
    const long bh = base0 + (long)h * 320 * 320;
    for (int q = 0; q < 64; ++q) {
      const long idx = bh + (long)q * 320;
      acc += cr[idx] + ct[idx];
    }
  }
  score[b * 256 + s] = acc * (1.0f / 768.0f);
}

__global__ __launch_bounds__(256) void ce_topk(
    const float* __restrict__ score, const int* __restrict__ gidx_s,
    int* __restrict__ topk, float* __restrict__ out_keep,
    float* __restrict__ out_rem) {
  __shared__ float sc[256];
  __shared__ int id[256];
  const int b = blockIdx.x, t = threadIdx.x;
  sc[t] = score[b * 256 + t];
  id[t] = t;
  __syncthreads();
  // Bitonic sort, descending by score, ties by ascending index (stable-like).
  for (int k = 2; k <= 256; k <<= 1) {
    for (int j = k >> 1; j > 0; j >>= 1) {
      const int ixj = t ^ j;
      if (ixj > t) {
        const float sa = sc[t], sb = sc[ixj];
        const int ia = id[t], ib = id[ixj];
        const bool aFirst = (sa > sb) || (sa == sb && ia < ib);
        const bool asc = ((t & k) == 0);
        if (asc ? !aFirst : aFirst) {
          sc[t] = sb;
          sc[ixj] = sa;
          id[t] = ib;
          id[ixj] = ia;
        }
      }
      __syncthreads();
    }
  }
  const int src = id[t];
  if (t < 180) {
    topk[b * 180 + t] = src;
    out_keep[b * 180 + t] = (float)gidx_s[b * 256 + src];
  } else {
    out_rem[b * 76 + (t - 180)] = (float)gidx_s[b * 256 + src];
  }
}

__global__ __launch_bounds__(256) void gather_tokens(
    const float* __restrict__ xres, const int* __restrict__ topk,
    float* __restrict__ out) {
  const int row = blockIdx.x;  // 0 .. B*244-1
  const int b = row / 244, i = row % 244;
  const int src = (i < 64) ? i : (64 + topk[b * 180 + (i - 64)]);
  const float* p = xres + ((long)b * 320 + src) * 768;
  float* o = out + (long)row * 768;
  const int t = threadIdx.x;
  o[t] = p[t];
  o[t + 256] = p[t + 256];
  o[t + 512] = p[t + 512];
}

__global__ void copy_idx_f(const int* __restrict__ s, float* __restrict__ d,
                           int n) {
  const int i = blockIdx.x * blockDim.x + threadIdx.x;
  if (i < n) d[i] = (float)s[i];
}

// ---------------------------------------------------------------------------

extern "C" void kernel_launch(void* const* d_in, const int* in_sizes, int n_in,
                              void* d_out, int out_size, void* d_ws,
                              size_t ws_size, hipStream_t stream) {
  (void)in_sizes; (void)n_in; (void)out_size; (void)ws_size;

  constexpr int Bb = 64, Ntok = 320, Hh = 12, HDd = 64;
  constexpr int M1 = Bb * Ntok;   // 20480 tokens (pre-keep)
  constexpr int NK = 244;         // 64 + ceil(0.7*256)
  constexpr int M2 = Bb * NK;     // 15616 tokens (post-keep)
  const float SCALEf = 0.125f;    // 64^-0.5

  const float* x_rgb  = (const float*)d_in[0];
  const float* x_tir  = (const float*)d_in[1];
  const int*   gidx_t = (const int*)d_in[2];
  const int*   gidx_s = (const int*)d_in[3];
  const float* n1g = (const float*)d_in[4];
  const float* n1b = (const float*)d_in[5];
  const float* qkv_w = (const float*)d_in[6];
  const float* qkv_b = (const float*)d_in[7];
  const float* proj_w = (const float*)d_in[8];
  const float* proj_b = (const float*)d_in[9];
  const float* n2g = (const float*)d_in[10];
  const float* n2b = (const float*)d_in[11];
  const float* fc1_w = (const float*)d_in[12];
  const float* fc1_b = (const float*)d_in[13];
  const float* fc2_w = (const float*)d_in[14];
  const float* fc2_b = (const float*)d_in[15];

  float* out = (float*)d_out;
  float* out_xrgb = out;                       // B*244*768
  float* out_xtir = out + 11993088;
  float* out_git  = out + 23986176;            // B*64
  float* out_gis  = out + 23990272;            // B*180
  float* out_rem  = out + 24001792;            // B*76
  float* corr_rgb = out + 24006656;            // B*12*320*320
  float* corr_tir = out + 102649856;

  // Workspace carve-up (bf16 region first, then f32).
  bf16_t* wqkv    = (bf16_t*)d_ws;
  bf16_t* wproj   = wqkv + (size_t)768 * 2304;
  bf16_t* wfc1    = wproj + (size_t)768 * 768;
  bf16_t* wfc2    = wfc1 + (size_t)768 * 3072;
  bf16_t* lnbuf   = wfc2 + (size_t)3072 * 768;
  bf16_t* qkvbuf  = lnbuf + (size_t)M1 * 768;
  bf16_t* attnbf  = qkvbuf + (size_t)M1 * 2304;
  bf16_t* attnout = attnbf + (size_t)Bb * 12 * 320 * 320;
  bf16_t* hbuf    = attnout + (size_t)M1 * 768;
  float*  xres0   = (float*)(hbuf + (size_t)M2 * 3072);
  float*  xres1   = xres0 + (size_t)M1 * 768;
  float*  score   = xres1 + (size_t)M1 * 768;
  int*    topk    = (int*)(score + Bb * 256);

  // Weights -> bf16 (deterministic per call).
  cvt_f32_bf16<<<(768L * 2304 + 255) / 256, 256, 0, stream>>>(qkv_w, wqkv, 768L * 2304);
  cvt_f32_bf16<<<(768L * 768 + 255) / 256, 256, 0, stream>>>(proj_w, wproj, 768L * 768);
  cvt_f32_bf16<<<(768L * 3072 + 255) / 256, 256, 0, stream>>>(fc1_w, wfc1, 768L * 3072);
  cvt_f32_bf16<<<(3072L * 768 + 255) / 256, 256, 0, stream>>>(fc2_w, wfc2, 3072L * 768);

  for (int m = 0; m < 2; ++m) {
    const float* x = m ? x_tir : x_rgb;
    float* corr = m ? corr_tir : corr_rgb;
    float* xres = m ? xres1 : xres0;

    layernorm_bf16<<<M1, 256, 0, stream>>>(x, n1g, n1b, lnbuf);

    // QKV: [20480x768] @ [768x2304] + b -> bf16
    gemm_wmma_bf16<EPI_PLAIN, false, bf16_t><<<dim3(18, 160, 1), 256, 0, stream>>>(
        lnbuf, wqkv, qkv_b, nullptr, qkvbuf, M1, 2304, 768, 768, 2304, 2304,
        1, 0, 0, 0, 0, 0, 0, 1.0f);

    // Scores: per (b,h): Q[320x64] @ K^T[64x320] * scale -> corr (f32)
    gemm_wmma_bf16<EPI_PLAIN, true, float><<<dim3(3, 3, Bb * Hh), 256, 0, stream>>>(
        qkvbuf, qkvbuf + 768, nullptr, nullptr, corr, Ntok, Ntok, HDd,
        2304, 2304, Ntok, Hh, 320L * 2304, 64, 320L * 2304, 64,
        (long)Hh * 320 * 320, 320L * 320, SCALEf);

    softmax_row<<<Bb * Hh * Ntok, 128, 0, stream>>>(corr, attnbf);

    // attn @ V: per (b,h): [320x320] @ [320x64] -> attnout[b,n,h*64+d] (bf16)
    gemm_wmma_bf16<EPI_PLAIN, false, bf16_t><<<dim3(1, 3, Bb * Hh), 256, 0, stream>>>(
        attnbf, qkvbuf + 1536, nullptr, nullptr, attnout, Ntok, HDd, Ntok,
        Ntok, 2304, 768, Hh, 12L * 320 * 320, 320L * 320, 320L * 2304, 64,
        320L * 768, 64, 1.0f);

    // proj + residual: x + (attnout @ Wp + bp) -> xres (f32)
    gemm_wmma_bf16<EPI_RES, false, float><<<dim3(6, 160, 1), 256, 0, stream>>>(
        attnout, wproj, proj_b, x, xres, M1, 768, 768, 768, 768, 768,
        1, 0, 0, 0, 0, 0, 0, 1.0f);
  }

  // Candidate elimination.
  ce_score<<<Bb, 256, 0, stream>>>(corr_rgb, corr_tir, score);
  ce_topk<<<Bb, 256, 0, stream>>>(score, gidx_s, topk, out_gis, out_rem);
  copy_idx_f<<<(4096 + 255) / 256, 256, 0, stream>>>(gidx_t, out_git, 4096);

  gather_tokens<<<Bb * NK, 256, 0, stream>>>(xres0, topk, out_xrgb);
  gather_tokens<<<Bb * NK, 256, 0, stream>>>(xres1, topk, out_xtir);

  // MLP with residual, in place on the gathered tokens.
  for (int m = 0; m < 2; ++m) {
    float* ox = m ? out_xtir : out_xrgb;
    layernorm_bf16<<<M2, 256, 0, stream>>>(ox, n2g, n2b, lnbuf);
    gemm_wmma_bf16<EPI_GELU, false, bf16_t><<<dim3(24, 122, 1), 256, 0, stream>>>(
        lnbuf, wfc1, fc1_b, nullptr, hbuf, M2, 3072, 768, 768, 3072, 3072,
        1, 0, 0, 0, 0, 0, 0, 1.0f);
    gemm_wmma_bf16<EPI_RES, false, float><<<dim3(6, 122, 1), 256, 0, stream>>>(
        hbuf, wfc2, fc2_b, ox, ox, M2, 768, 3072, 3072, 768, 768,
        1, 0, 0, 0, 0, 0, 0, 1.0f);
  }
}
